// MultiHeadAttention_44212393345819
// MI455X (gfx1250) — matrix operand order repaired and checked by reading, not compiled, not run on hardware
//
#include <hip/hip_runtime.h>
#include <stdint.h>

// ---------------------------------------------------------------------------
// MHA with dense edge bias, MI455X (gfx1250), wave32 WMMA f16 + TDM staging.
// B=8 N=1024 HID=256 H=8 D=32 EF=16 E=262144
// ---------------------------------------------------------------------------

#define BB   8
#define NN_  1024
#define HID_ 256
#define HH   8
#define DD   32
#define EF_  16
#define EE   262144

typedef _Float16 half_t;
typedef __attribute__((ext_vector_type(16))) _Float16      v16h;
typedef __attribute__((ext_vector_type(8)))  float         v8f;
typedef __attribute__((ext_vector_type(4)))  unsigned int  v4u;
typedef __attribute__((ext_vector_type(8)))  int           v8i_;
typedef __attribute__((ext_vector_type(4)))  int           v4i;

__device__ __forceinline__ v8f wmma_f16(v16h a, v16h b, v8f c) {
    return __builtin_amdgcn_wmma_f32_16x16x32_f16(
        /*neg_a=*/false, a, /*neg_b=*/false, b,
        /*c_mod=*/(short)0, c, /*reuse_a=*/false, /*reuse_b=*/false);
}

// ---------------------------------------------------------------------------
// TDM: 1-D contiguous DMA of `nelem` f16 elements global -> LDS.
// D# per CDNA5 ISA 8.3/8.4: group0 = {count=1, lds_addr, global_addr, type=2},
// group1 = {data_size=2B, tensor_dim0=tile_dim0=nelem (single row),
//           tensor_dim1=tile_dim1=1, stride0=nelem}.  Groups 2/3 unused (2D).
// Tracked on TENSORcnt; EXEC is ignored by TDM (issue from one wave only).
// This toolchain's builtin takes 6 args (g0, g1, g2, g3, spare, cpol).
// ---------------------------------------------------------------------------
__device__ __forceinline__ void tdm_load_1d_f16(uint32_t lds_off,
                                                const half_t* gptr,
                                                uint32_t nelem /* <= 65535 */)
{
    const uint64_t ga = (uint64_t)(uintptr_t)gptr;
    v4u g0;
    g0[0] = 1u;                                        // count=1, user D#
    g0[1] = lds_off;                                   // lds_addr (bytes)
    g0[2] = (uint32_t)(ga & 0xFFFFFFFFu);              // global_addr[31:0]
    g0[3] = (uint32_t)((ga >> 32) & 0x01FFFFFFu)       // global_addr[56:32]
          | (2u << 30);                                // type=2 ("image")
    v8i_ g1;
    g1[0] = (int)(1u << 16);                           // data_size=1 (2 bytes)
    g1[1] = (int)((nelem & 0xFFFFu) << 16);            // tensor_dim0[15:0]
    g1[2] = (int)(((nelem >> 16) & 0xFFFFu)            // tensor_dim0[31:16]
          | (1u << 16));                               // tensor_dim1 = 1
    g1[3] = (int)((nelem & 0xFFFFu) << 16);            // tile_dim0 = nelem
    g1[4] = 1;                                         // tile_dim1=1, tile_dim2=0
    g1[5] = (int)nelem;                                // tensor_dim0_stride lo
    g1[6] = 0;                                         // stride hi, dim1_stride lo
    g1[7] = 0;
    const v4i  z4 = {};
    const v8i_ z8 = {};
    __builtin_amdgcn_tensor_load_to_lds(g0, g1, z4, z4, z8, 0);
}

// ---------------------------------------------------------------------------
// Kernel 1: Y[m,c] = (sum_k X[m,k]*W[c,k] + bias[c]) * scale
// X: (8192,256) f32, W: (256,256) f32.  Output packed (B,H,N,D) f16.
// One wave per 16x16 tile; K looped 8 x 32 (WMMA f16, K=32).
// ---------------------------------------------------------------------------
__global__ __launch_bounds__(256)
void proj_kernel(const float* __restrict__ X, const float* __restrict__ W,
                 const float* __restrict__ bvec, half_t* __restrict__ out,
                 float scale)
{
    const int wave  = threadIdx.x >> 5;
    const int lane  = threadIdx.x & 31;
    const int lrow  = lane & 15;
    const int lhalf = lane >> 4;

    const int n0 = (blockIdx.x * 8 + wave) * 16;   // output col tile
    const int m0 = blockIdx.y * 16;                // output row tile
    const int arow  = m0 + lrow;
    const int abase = lhalf * 8;                   // ISA A-layout K offset

    v8f acc = {};
    for (int kc = 0; kc < HID_; kc += 32) {
        v16h a;
        const float* xp = X + (size_t)arow * HID_ + kc + abase;
        #pragma unroll
        for (int g = 0; g < 2; ++g)
            #pragma unroll
            for (int w = 0; w < 8; ++w)
                a[g * 8 + w] = (half_t)xp[16 * g + w];
        v16h b;
        const float* wp = W + (size_t)(n0 + lrow) * HID_ + kc + lhalf * 8;
        #pragma unroll
        for (int i = 0; i < 16; ++i) b[i] = (half_t)wp[i + (i & 8)];
        acc = wmma_f16(a, b, acc);
    }

    const int col = n0 + lrow;
    const float bv = bvec[col];
    const int h = col >> 5, d = col & 31;
    #pragma unroll
    for (int r = 0; r < 8; ++r) {
        const int grow = m0 + r + 8 * lhalf;       // global row (b*N+n)
        const int bi = grow >> 10, nn = grow & (NN_ - 1);
        const float v = (acc[r] + bv) * scale;
        out[(((size_t)(bi * HH + h) * NN_) + nn) * DD + d] = (half_t)v;
    }
}

// ---------------------------------------------------------------------------
// Kernel 2: per-edge head weights + scatter-add into dense bias (B,H,N,N) f32
// ---------------------------------------------------------------------------
__global__ __launch_bounds__(256)
void edge_bias_kernel(const float* __restrict__ edge_attr,
                      const int* __restrict__ edge_index,
                      const int* __restrict__ batch,
                      const float* __restrict__ We, const float* __restrict__ be,
                      float* __restrict__ bias)
{
    const int e = blockIdx.x * blockDim.x + threadIdx.x;
    if (e >= EE) return;
    float ea[EF_];
    #pragma unroll
    for (int k = 0; k < EF_; ++k) ea[k] = edge_attr[(size_t)e * EF_ + k];
    const int src = edge_index[e];
    const int dst = edge_index[EE + e];
    const int g   = batch[src];
    #pragma unroll
    for (int h = 0; h < HH; ++h) {
        float v = be[h];
        #pragma unroll
        for (int k = 0; k < EF_; ++k) v += ea[k] * We[h * EF_ + k];
        atomicAdd(&bias[(((size_t)(g * HH + h) * NN_) + src) * NN_ + dst], v);
    }
}

// ---------------------------------------------------------------------------
// Kernel 3: flash attention.  One block = one (b,h) x 128 query rows.
// K/V tiles (64KB each, contiguous in (B,H,N,D)) DMA'd to LDS by the TDM,
// fenced with s_wait_tensorcnt.  Per 32-key chunk: 2 score WMMAs (K=D=32),
// NT bias+mask loads, online softmax, P via LDS (D->A layout), 2 PV WMMAs.
// grid = (8, 64), block = 256 (8 waves).  LDS: 64+64+8 = 136 KB.
// ---------------------------------------------------------------------------
__global__ __launch_bounds__(256)
void attn_kernel(const half_t* __restrict__ qh, const half_t* __restrict__ kh,
                 const half_t* __restrict__ vh, const float* __restrict__ bias,
                 const unsigned char* __restrict__ mask,
                 half_t* __restrict__ attn_out)
{
    __shared__ alignas(128) half_t Klds[NN_ * DD];   // 64 KB
    __shared__ alignas(128) half_t Vlds[NN_ * DD];   // 64 KB
    __shared__ half_t Plds[8][16][32];               // 8 KB

    const int wave  = threadIdx.x >> 5;
    const int lane  = threadIdx.x & 31;
    const int lrow  = lane & 15;
    const int lhalf = lane >> 4;

    const int bh = blockIdx.y;
    const int b  = bh >> 3, h = bh & 7;
    const int m0 = (blockIdx.x * 8 + wave) * 16;     // query row tile

    const half_t* qb = qh + (size_t)bh * NN_ * DD;
    const half_t* kb = kh + (size_t)bh * NN_ * DD;
    const half_t* vb = vh + (size_t)bh * NN_ * DD;
    const float*  bb = bias + (size_t)bh * NN_ * NN_;
    const unsigned char* mb = mask + (size_t)b * NN_ * NN_;

    // Kick off TDM loads of the whole K and V tiles for this (b,h).
    if (wave == 0) {
        tdm_load_1d_f16((uint32_t)(uintptr_t)&Klds[0], kb, NN_ * DD);
        tdm_load_1d_f16((uint32_t)(uintptr_t)&Vlds[0], vb, NN_ * DD);
    }

    // Q tile in A layout (global, overlapped with the DMA).
    v16h aq;
    {
        const half_t* qr = qb + (size_t)(m0 + lrow) * DD;
        const int abase = lhalf * 8;
        #pragma unroll
        for (int g = 0; g < 2; ++g)
            #pragma unroll
            for (int w = 0; w < 8; ++w)
                aq[g * 8 + w] = qr[abase + 16 * g + w];
    }

    if (wave == 0) __builtin_amdgcn_s_wait_tensorcnt(0);
    __syncthreads();

    v8f acc0 = {}, acc1 = {};
    float mrun[8], lrun[8];
    #pragma unroll
    for (int r = 0; r < 8; ++r) { mrun[r] = -3.0e38f; lrun[r] = 0.0f; }

    for (int key0 = 0; key0 < NN_; key0 += 32) {
        // K tiles in B layout from LDS: b[i] = K[key0(+16)+lrow][lhalf*8+i+(i&8)]
        v16h bk0, bk1;
        {
            const half_t* kr0 = &Klds[(key0 + lrow) * DD + lhalf * 8];
            const half_t* kr1 = kr0 + 16 * DD;
            #pragma unroll
            for (int i = 0; i < 16; ++i) { bk0[i] = kr0[i + (i & 8)]; bk1[i] = kr1[i + (i & 8)]; }
        }

        const v8f z = {};
        v8f s0 = wmma_f16(aq, bk0, z);     // cols key0 .. key0+15
        v8f s1 = wmma_f16(aq, bk1, z);     // cols key0+16 .. key0+31

        // bias + mask + online softmax per row (row = r + 8*lhalf).
        // Bias/mask are streamed exactly once (268 MB) -> non-temporal loads
        // so the stream does not evict K/V/W working set from L2 (192 MB).
        #pragma unroll
        for (int r = 0; r < 8; ++r) {
            const int grow = m0 + r + 8 * lhalf;
            const float* bp = bb + (size_t)grow * NN_ + key0;
            const unsigned char* mp = mb + (size_t)grow * NN_ + key0;
            float x0 = s0[r] + __builtin_nontemporal_load(bp + lrow);
            float x1 = s1[r] + __builtin_nontemporal_load(bp + 16 + lrow);
            if (__builtin_nontemporal_load(mp + lrow))      x0 = -1.0e9f;
            if (__builtin_nontemporal_load(mp + 16 + lrow)) x1 = -1.0e9f;

            float mx = fmaxf(x0, x1);
            #pragma unroll
            for (int off = 1; off < 16; off <<= 1)
                mx = fmaxf(mx, __shfl_xor(mx, off, 32));   // within 16-lane half
            const float mnew  = fmaxf(mrun[r], mx);
            const float alpha = __expf(mrun[r] - mnew);
            const float p0 = __expf(x0 - mnew);
            const float p1 = __expf(x1 - mnew);
            float rs = p0 + p1;
            #pragma unroll
            for (int off = 1; off < 16; off <<= 1)
                rs += __shfl_xor(rs, off, 32);
            lrun[r] = lrun[r] * alpha + rs;
            mrun[r] = mnew;
            acc0[r] *= alpha;
            acc1[r] *= alpha;

            Plds[wave][r + 8 * lhalf][lrow]      = (half_t)p0;
            Plds[wave][r + 8 * lhalf][16 + lrow] = (half_t)p1;
        }

        // Same-wave LDS is in-order; fence compiler + dscnt before re-read.
        __builtin_amdgcn_wave_barrier();
        asm volatile("s_wait_dscnt 0" ::: "memory");

        // P in A layout (16 rows x 32 local keys).
        v16h ap;
        {
            const int abase = lhalf * 8;
            #pragma unroll
            for (int g = 0; g < 2; ++g)
                #pragma unroll
                for (int w = 0; w < 8; ++w)
                    ap[g * 8 + w] = Plds[wave][lrow][abase + 16 * g + w];
        }
        // V tiles in B layout from LDS: b[i] = V[key0+lhalf*8+i+(i&8)][d]
        v16h bv0, bv1;
        {
            const half_t* vr = &Vlds[(key0 + lhalf * 8) * DD];
            #pragma unroll
            for (int i = 0; i < 16; ++i) {
                bv0[i] = vr[(i + (i & 8)) * DD + lrow];
                bv1[i] = vr[(i + (i & 8)) * DD + 16 + lrow];
            }
        }
        acc0 = wmma_f16(ap, bv0, acc0);
        acc1 = wmma_f16(ap, bv1, acc1);

        __builtin_amdgcn_wave_barrier();   // keep Plds reuse ordered vs next iter
    }

    // epilogue: out[b][n][h*32+d] = acc/l   (f16, consumed by output GEMM)
    #pragma unroll
    for (int r = 0; r < 8; ++r) {
        const int grow = m0 + r + 8 * lhalf;
        const float inv = 1.0f / lrun[r];
        half_t* op = attn_out + ((size_t)(b * NN_ + grow)) * HID_ + h * DD;
        op[lrow]      = (half_t)(acc0[r] * inv);
        op[16 + lrow] = (half_t)(acc1[r] * inv);
    }
}

// ---------------------------------------------------------------------------
// Kernel 4: output projection  out[m,c] = sum_k A[m,k]*Wo[c,k] + bo[c]  (f32)
// ---------------------------------------------------------------------------
__global__ __launch_bounds__(256)
void out_proj_kernel(const half_t* __restrict__ A, const float* __restrict__ W,
                     const float* __restrict__ bvec, float* __restrict__ out)
{
    const int wave  = threadIdx.x >> 5;
    const int lane  = threadIdx.x & 31;
    const int lrow  = lane & 15;
    const int lhalf = lane >> 4;

    const int n0 = (blockIdx.x * 8 + wave) * 16;
    const int m0 = blockIdx.y * 16;
    const int arow  = m0 + lrow;
    const int abase = lhalf * 8;

    v8f acc = {};
    for (int kc = 0; kc < HID_; kc += 32) {
        v16h a;
        const half_t* ap = A + (size_t)arow * HID_ + kc + abase;
        #pragma unroll
        for (int g = 0; g < 2; ++g)
            #pragma unroll
            for (int w = 0; w < 8; ++w)
                a[g * 8 + w] = ap[16 * g + w];
        v16h bm;
        const float* wp = W + (size_t)(n0 + lrow) * HID_ + kc + lhalf * 8;
        #pragma unroll
        for (int i = 0; i < 16; ++i) bm[i] = (half_t)wp[i + (i & 8)];
        acc = wmma_f16(a, bm, acc);
    }

    const int col = n0 + lrow;
    const float bv = bvec[col];
    #pragma unroll
    for (int r = 0; r < 8; ++r) {
        const int grow = m0 + r + 8 * lhalf;
        out[(size_t)grow * HID_ + col] = acc[r] + bv;
    }
}

// ---------------------------------------------------------------------------
// Launch
// ---------------------------------------------------------------------------
extern "C" void kernel_launch(void* const* d_in, const int* in_sizes, int n_in,
                              void* d_out, int out_size, void* d_ws, size_t ws_size,
                              hipStream_t stream)
{
    const float* q         = (const float*)d_in[0];
    const float* k         = (const float*)d_in[1];
    const float* v         = (const float*)d_in[2];
    const float* edge_attr = (const float*)d_in[3];
    const int*   edge_idx  = (const int*)  d_in[4];
    const int*   batch     = (const int*)  d_in[5];
    const unsigned char* attn_mask = (const unsigned char*)d_in[6]; // bool bytes
    const float* Wq = (const float*)d_in[7];
    const float* bq = (const float*)d_in[8];
    const float* Wk = (const float*)d_in[9];
    const float* bk = (const float*)d_in[10];
    const float* Wv = (const float*)d_in[11];
    const float* bv = (const float*)d_in[12];
    const float* We = (const float*)d_in[13];
    const float* be = (const float*)d_in[14];
    const float* Wo = (const float*)d_in[15];
    const float* bo = (const float*)d_in[16];
    float* out = (float*)d_out;

    // Workspace: dense bias (268 MB f32) + 4 x (8192*256) f16 activations
    char* ws = (char*)d_ws;
    const size_t biasBytes = (size_t)BB * HH * NN_ * NN_ * sizeof(float);
    const size_t actElems  = (size_t)BB * NN_ * HID_;
    float*  bias = (float*)ws;
    half_t* qh   = (half_t*)(ws + biasBytes);
    half_t* kh   = qh + actElems;
    half_t* vh   = kh + actElems;
    half_t* ao   = vh + actElems;

    (void)hipMemsetAsync(bias, 0, biasBytes, stream);

    const dim3 pgrid(2, 512);                      // 16 N-tiles x 512 M-tiles
    const float scale = 0.17677669529663687f;      // D^-0.5, D=32
    proj_kernel<<<pgrid, 256, 0, stream>>>(q, Wq, bq, qh, scale);
    proj_kernel<<<pgrid, 256, 0, stream>>>(k, Wk, bk, kh, 1.0f);
    proj_kernel<<<pgrid, 256, 0, stream>>>(v, Wv, bv, vh, 1.0f);

    edge_bias_kernel<<<dim3(EE / 256), 256, 0, stream>>>(
        edge_attr, edge_idx, batch, We, be, bias);

    attn_kernel<<<dim3(8, 64), 256, 0, stream>>>(qh, kh, vh, bias, attn_mask, ao);

    out_proj_kernel<<<pgrid, 256, 0, stream>>>(ao, Wo, bo, out);
}
